// roPEAttentionHead_56324201120209
// MI455X (gfx1250) — matrix operand order repaired and checked by reading, not compiled, not run on hardware
//
#include <hip/hip_runtime.h>
#include <hip/hip_bf16.h>

// ---------------------------------------------------------------------------
// RoPE attention head for MI455X (gfx1250, wave32, WMMA bf16 16x16x32)
//   B=16, C=2048, D=256
// Phase 1: fused QKV projection + rotary embedding -> bf16 workspace
// Phase 2: causal flash-attention (online softmax), K/V tiles staged to LDS
//          by the Tensor Data Mover (TENSOR_LOAD_TO_LDS, TENSORcnt) -> fp32
// ---------------------------------------------------------------------------

typedef __attribute__((ext_vector_type(16))) __bf16 v16bf;
typedef __attribute__((ext_vector_type(8)))  __bf16 v8bf;
typedef __attribute__((ext_vector_type(8)))  float  v8f;
typedef __attribute__((ext_vector_type(4)))  unsigned int u32x4;
typedef __attribute__((ext_vector_type(8)))  int i32x8;
typedef __attribute__((ext_vector_type(4)))  int i32x4;

#define Bx   16
#define Cx   2048
#define Dx   256

static __device__ __forceinline__ v16bf mk16(v8bf lo, v8bf hi) {
    v16bf r;
#pragma unroll
    for (int i = 0; i < 8; ++i) { r[i] = lo[i]; r[i + 8] = hi[i]; }
    return r;
}

static __device__ __forceinline__ v8f wmma_bf16(v16bf a, v16bf b, v8f c) {
    // D(f32 16x16) = A(bf16 16x32) * B(bf16 32x16) + C
    return __builtin_amdgcn_wmma_f32_16x16x32_bf16(
        /*neg_a=*/false, a, /*neg_b=*/false, b,
        /*c_mod=*/(short)0, c, /*reuse_a=*/false, /*reuse_b=*/false);
}

// ---------------------------------------------------------------------------
// Tensor Data Mover: 2D tile load (bf16 elements) global -> LDS.
// Builds the D# descriptor per CDNA5 ISA ch.8 (group0 128b + group1 256b,
// groups 2/3 zero for a 2D tensor) and issues TENSOR_LOAD_TO_LDS.
// Caller must be wave-uniform; completion via s_wait_tensorcnt.
// ---------------------------------------------------------------------------
static __device__ __forceinline__ void tdm_load_2d_bf16(
    unsigned int lds_off, const __bf16* gptr,
    unsigned int tile_d0, unsigned int tile_d1,
    unsigned int tensor_d0, unsigned int tensor_d1,
    unsigned int stride_d0)
{
    const unsigned long long ga = (unsigned long long)(size_t)gptr;
    u32x4 g0;
    g0[0] = 1u;                                   // count=1, user descriptor
    g0[1] = lds_off;                              // lds_addr (bytes)
    g0[2] = (unsigned int)(ga & 0xffffffffu);     // global_addr[31:0]
    g0[3] = (unsigned int)((ga >> 32) & 0x01ffffffu) | (2u << 30); // addr[56:32], type=2
    i32x8 g1;
    g1[0] = (int)(1u << 16);                      // wg_mask=0, data_size=1 (2B)
    g1[1] = (int)(tensor_d0 << 16);               // tensor_dim0[15:0]  @ [31:16]
    g1[2] = (int)((tensor_d0 >> 16) | (tensor_d1 << 16)); // dim0 hi | dim1 lo
    g1[3] = (int)((tensor_d1 >> 16) | (tile_d0 << 16));   // dim1 hi | tile_dim0
    g1[4] = (int)(tile_d1 & 0xffffu);             // tile_dim1 (tile_dim2=0)
    g1[5] = (int)stride_d0;                       // tensor_dim0_stride[31:0]
    g1[6] = 0;                                    // stride0 hi | dim1_stride lo
    g1[7] = 0;                                    // dim1_stride hi
    const i32x4 z4 = {0, 0, 0, 0};
#if __clang_major__ >= 23
    const i32x8 z8 = {0, 0, 0, 0, 0, 0, 0, 0};
    __builtin_amdgcn_tensor_load_to_lds(g0, g1, z4, z4, z8, 0);
#else
    __builtin_amdgcn_tensor_load_to_lds(g0, g1, z4, z4, 0);
#endif
}

// ---------------------------------------------------------------------------
// Kernel 1: Q/K/V projection (x @ W^T + b) with RoPE applied to Q and K.
// Grid: (B*C/16) blocks of 256 threads (8 waves).
// ---------------------------------------------------------------------------
__global__ __launch_bounds__(256) void qkv_rope_kernel(
    const float* __restrict__ x,
    const float* __restrict__ Wq, const float* __restrict__ bq,
    const float* __restrict__ Wk, const float* __restrict__ bk,
    const float* __restrict__ Wv, const float* __restrict__ bv,
    const float* __restrict__ R,
    __bf16* __restrict__ Qr, __bf16* __restrict__ Kr, __bf16* __restrict__ Vr)
{
    __shared__ __align__(16) __bf16 Xs[16][Dx];   // 8 KB bf16 x tile

    const int tid = threadIdx.x;
    const int rt  = blockIdx.x;                   // 16-row tile over B*C

    // Stage x tile -> bf16 LDS (each thread converts 16 elements)
    {
        const int row = tid >> 4;                 // 0..15
        const int cb  = (tid & 15) * 16;          // 16 columns
        const float* src = x + ((size_t)(rt * 16 + row)) * Dx + cb;
#pragma unroll
        for (int i = 0; i < 16; ++i) Xs[row][cb + i] = (__bf16)src[i];
    }
    __syncthreads();

    const int lane = tid & 31;
    const int wave = tid >> 5;
    const int am   = lane & 15;                   // A-fragment row (M)
    const int kh   = lane >> 4;                   // A-fragment K-half

    // Resident A fragments: 8 K-steps of 32
    v16bf afrag[8];
#pragma unroll
    for (int s = 0; s < 8; ++s) {
        v8bf lo = *(const v8bf*)&Xs[am][s * 32 + kh * 8];
        v8bf hi = *(const v8bf*)&Xs[am][s * 32 + 16 + kh * 8];
        afrag[s] = mk16(lo, hi);
    }

    const int nn = lane & 15;                     // D-layout column within tile
    const int mh = lane >> 4;                     // D-layout row-half

    for (int ci = 0; ci < 6; ++ci) {
        const int ct    = wave * 6 + ci;          // 0..47
        const int mat   = ct >> 4;                // 0=Q 1=K 2=V
        const int ntile = ct & 15;
        const float* W   = (mat == 0) ? Wq : (mat == 1) ? Wk : Wv;
        const float* bia = (mat == 0) ? bq : (mat == 1) ? bk : bv;
        __bf16*      dst = (mat == 0) ? Qr : (mat == 1) ? Kr : Vr;

        v8f acc;
#pragma unroll
        for (int i = 0; i < 8; ++i) acc[i] = 0.0f;

#pragma unroll
        for (int s = 0; s < 8; ++s) {
            const int kg = s * 32 + kh * 8;       // B-fragment: lane = N (W row)
            v16bf bfrag;
#pragma unroll
            for (int e = 0; e < 16; ++e)          // element = K
                bfrag[e] = (__bf16)W[(size_t)(ntile * 16 + nn) * Dx + kg + (e & 7) + ((e >> 3) << 4)];
            acc = wmma_bf16(afrag[s], bfrag, acc);
        }

        const int   col = ntile * 16 + nn;
        const float bcv = bia[col];

        if (mat < 2) {
            // RoPE: pair partner is the adjacent lane (col ^ 1)
            const int   p   = col >> 1;
            const float sgn = (col & 1) ? 1.0f : -1.0f;
#pragma unroll
            for (int r = 0; r < 8; ++r) {
                float self = acc[r] + bcv;
                float part = __shfl_xor(self, 1, 32);
                const int    row = rt * 16 + r + 8 * mh;
                const int    pos = row & (Cx - 1);
                const size_t rb  = (size_t)pos * (Dx * Dx);
                const float  cv  = R[rb + (size_t)(2 * p) * Dx + 2 * p];
                const float  sv  = R[rb + (size_t)(2 * p + 1) * Dx + 2 * p];
                const float  o   = cv * self + sgn * sv * part;
                dst[(size_t)row * Dx + col] = (__bf16)o;
            }
        } else {
#pragma unroll
            for (int r = 0; r < 8; ++r) {
                const int row = rt * 16 + r + 8 * mh;
                dst[(size_t)row * Dx + col] = (__bf16)(acc[r] + bcv);
            }
        }
    }
}

// ---------------------------------------------------------------------------
// Kernel 2: causal flash-attention.
// Grid: (C/64, B), 256 threads = 8 waves: 4 query sub-tiles x 2 V-halves.
// K/V 32-key tiles DMA'd to LDS by the Tensor Data Mover (wave 0 issues,
// TENSORcnt completion, block barrier publishes). Uniform trip counts so
// block barriers are legal inside the loop.
// ---------------------------------------------------------------------------
__global__ __launch_bounds__(256) void attn_kernel(
    const __bf16* __restrict__ Qr, const __bf16* __restrict__ Kr,
    const __bf16* __restrict__ Vr, float* __restrict__ out)
{
    __shared__ __align__(16) __bf16 Ks[32][Dx];      // 16 KB
    __shared__ __align__(16) __bf16 Vs[32][Dx];      // 16 KB
    __shared__ __align__(16) __bf16 Ps[8][16][32];   // 8 KB (per-wave P tiles)

    const int tid  = threadIdx.x;
    const int lane = tid & 31;
    const int wave = tid >> 5;
    const int b    = blockIdx.y;
    const int q0   = blockIdx.x * 64;
    const int qsub = wave >> 1;
    const int vh   = wave & 1;                       // which half of V columns
    const int qw   = q0 + qsub * 16;
    const int nn   = lane & 15;
    const int mh   = lane >> 4;
    const size_t base = (size_t)b * Cx * Dx;

    const unsigned int ldsK = (unsigned int)(size_t)(&Ks[0][0]);
    const unsigned int ldsV = (unsigned int)(size_t)(&Vs[0][0]);

    // Resident rotated-Q fragments (A layout), 8 K-steps over D=256
    v16bf qfrag[8];
#pragma unroll
    for (int s = 0; s < 8; ++s) {
        const __bf16* qrow = Qr + base + (size_t)(qw + nn) * Dx;
        v8bf lo = *(const v8bf*)(qrow + s * 32 + mh * 8);
        v8bf hi = *(const v8bf*)(qrow + s * 32 + 16 + mh * 8);
        qfrag[s] = mk16(lo, hi);
    }

    v8f   O[8];
    float mi[8], li[8];
#pragma unroll
    for (int j = 0; j < 8; ++j) {
#pragma unroll
        for (int r = 0; r < 8; ++r) O[j][r] = 0.0f;
    }
#pragma unroll
    for (int r = 0; r < 8; ++r) { mi[r] = -__builtin_inff(); li[r] = 0.0f; }

    const float scale = 0.0625f;                     // 1/sqrt(256)
    const int   jjhi  = (q0 + 63) >> 5;              // inclusive, uniform in block

    for (int jj = 0; jj <= jjhi; ++jj) {
        // ---- TDM: DMA 32x256 bf16 K and V tiles into LDS (wave 0 only) ----
        if (wave == 0) {
            tdm_load_2d_bf16(ldsK, Kr + base + (size_t)(jj * 32) * Dx,
                             /*tile*/ Dx, 32, /*tensor*/ Dx, Cx, /*stride*/ Dx);
            tdm_load_2d_bf16(ldsV, Vr + base + (size_t)(jj * 32) * Dx,
                             /*tile*/ Dx, 32, /*tensor*/ Dx, Cx, /*stride*/ Dx);
            __builtin_amdgcn_s_wait_tensorcnt(0);
        }
        if (jj < jjhi) {                             // prefetch next tiles to L2
            __builtin_prefetch(Kr + base + (size_t)((jj + 1) * 32) * Dx + lane * 16, 0, 1);
            __builtin_prefetch(Vr + base + (size_t)((jj + 1) * 32) * Dx + lane * 16, 0, 1);
        }
        __syncthreads();

        if (jj * 32 <= qw + 15) {                    // causal: this wave needs tile
            // ---- S = Qrot . Krot^T for 32 keys (two 16-key WMMA tiles) ----
            v8f S0, S1;
#pragma unroll
            for (int i = 0; i < 8; ++i) { S0[i] = 0.0f; S1[i] = 0.0f; }
#pragma unroll
            for (int s = 0; s < 8; ++s) {
                v16bf b0, b1;                        // B: lane = key, elem = d index
#pragma unroll
                for (int e = 0; e < 16; ++e) {
                    b0[e] = Ks[nn][s * 32 + mh * 8 + (e & 7) + ((e >> 3) << 4)];
                    b1[e] = Ks[16 + nn][s * 32 + mh * 8 + (e & 7) + ((e >> 3) << 4)];
                }
                S0 = wmma_bf16(qfrag[s], b0, S0);
                S1 = wmma_bf16(qfrag[s], b1, S1);
            }

            // ---- mask + online softmax (row stats across 16-lane groups) ----
            const int k0 = jj * 32 + nn;
            const int k1 = k0 + 16;
#pragma unroll
            for (int r = 0; r < 8; ++r) {
                const int q = qw + r + 8 * mh;
                float s0 = (k0 <= q) ? S0[r] * scale : -__builtin_inff();
                float s1 = (k1 <= q) ? S1[r] * scale : -__builtin_inff();
                float rm = fmaxf(s0, s1);
#pragma unroll
                for (int w = 1; w < 16; w <<= 1) rm = fmaxf(rm, __shfl_xor(rm, w, 32));
                const float mnew  = fmaxf(mi[r], rm);
                const float alpha = __expf(mi[r] - mnew);
                const float p0    = __expf(s0 - mnew);
                const float p1    = __expf(s1 - mnew);
                float rs = p0 + p1;
#pragma unroll
                for (int w = 1; w < 16; w <<= 1) rs += __shfl_xor(rs, w, 32);
                li[r] = li[r] * alpha + rs;
                mi[r] = mnew;
#pragma unroll
                for (int j = 0; j < 8; ++j) O[j][r] *= alpha;
                const int mrow = r + 8 * mh;
                Ps[wave][mrow][nn]      = (__bf16)p0;
                Ps[wave][mrow][16 + nn] = (__bf16)p1;
            }
            // wave-local LDS transpose fence (same-wave DS ops are in order;
            // block order + wait on the split DS counter)
            __builtin_amdgcn_wave_barrier();
            asm volatile("s_wait_dscnt 0" ::: "memory");

            // ---- O += P @ V (A = P 16x32, B = V 32x16 per column tile) ----
            v8bf plo = *(const v8bf*)&Ps[wave][nn][mh * 8];
            v8bf phi = *(const v8bf*)&Ps[wave][nn][16 + mh * 8];
            v16bf pfrag = mk16(plo, phi);
#pragma unroll
            for (int j = 0; j < 8; ++j) {
                const int nt = vh * 8 + j;
                v16bf vfrag;
#pragma unroll
                for (int e = 0; e < 16; ++e)
                    vfrag[e] = Vs[mh * 8 + (e & 7) + ((e >> 3) << 4)][nt * 16 + nn];
                O[j] = wmma_bf16(pfrag, vfrag, O[j]);
            }
        }
        __syncthreads();                             // before LDS tiles reused
    }

    // ---- normalize and store fp32 output ----
#pragma unroll
    for (int r = 0; r < 8; ++r) {
        const float inv = 1.0f / li[r];
        const int   q   = qw + r + 8 * mh;
#pragma unroll
        for (int j = 0; j < 8; ++j) {
            const int col = (vh * 8 + j) * 16 + nn;
            out[((size_t)b * Cx + q) * Dx + col] = O[j][r] * inv;
        }
    }
}

// ---------------------------------------------------------------------------
// Host launcher
// ---------------------------------------------------------------------------
extern "C" void kernel_launch(void* const* d_in, const int* in_sizes, int n_in,
                              void* d_out, int out_size, void* d_ws, size_t ws_size,
                              hipStream_t stream) {
    const float* x  = (const float*)d_in[0];
    const float* Wq = (const float*)d_in[1];
    const float* bq = (const float*)d_in[2];
    const float* Wk = (const float*)d_in[3];
    const float* bk = (const float*)d_in[4];
    const float* Wv = (const float*)d_in[5];
    const float* bv = (const float*)d_in[6];
    const float* R  = (const float*)d_in[7];

    const size_t elems = (size_t)Bx * Cx * Dx;       // 8.4M per tensor
    __bf16* Qr = (__bf16*)d_ws;                      // rotated Q (bf16)
    __bf16* Kr = Qr + elems;                         // rotated K (bf16)
    __bf16* Vr = Kr + elems;                         // V (bf16)

    qkv_rope_kernel<<<(Bx * Cx) / 16, 256, 0, stream>>>(
        x, Wq, bq, Wk, bk, Wv, bv, R, Qr, Kr, Vr);

    attn_kernel<<<dim3(Cx / 64, Bx), 256, 0, stream>>>(
        Qr, Kr, Vr, (float*)d_out);
}
